// GAT_1726576853703
// MI455X (gfx1250) — matrix operand order repaired and checked by reading, not compiled, not run on hardware
//
#include <hip/hip_runtime.h>
#include <hip/hip_bf16.h>

// ---------------------------------------------------------------------------
// GAT (2 layers) + MLP classifier for MI455X (gfx1250, wave32).
// Edge phase: coalesced gather/scatter with float atomics (memory bound,
// ~4 GB traffic -> HBM-limited at 23.3 TB/s).
// Dense phase: double-buffered LDS-staged GEMM on v_wmma_f32_16x16x32_f16.
// ---------------------------------------------------------------------------

#define NN 50000
#define NE 1600000
#define NG 64

typedef __attribute__((ext_vector_type(16))) _Float16 v16h;
typedef __attribute__((ext_vector_type(8)))  _Float16 v8h;
typedef __attribute__((ext_vector_type(4)))  _Float16 v4h;
typedef __attribute__((ext_vector_type(8)))  float    v8f;

// monotone float<->uint mapping so unsigned atomicMax == float max
__device__ __forceinline__ unsigned fmap(float f) {
    unsigned u = __float_as_uint(f);
    return (u & 0x80000000u) ? ~u : (u | 0x80000000u);
}
__device__ __forceinline__ float funmap(unsigned u) {
    return (u & 0x80000000u) ? __uint_as_float(u & 0x7FFFFFFFu)
                             : __uint_as_float(~u);
}

// ---- degree (node feature) ------------------------------------------------
__global__ void k_degree(const int* __restrict__ dst, float* __restrict__ deg, int E) {
    int e = blockIdx.x * blockDim.x + threadIdx.x;
    if (e < E) atomicAdd(&deg[dst[e]], 1.0f);
}

// ---- layer1: z1 = deg * W1 (K=1 outer product) + el/er reductions ---------
__global__ void k_l1_node(const float* __restrict__ deg, const float* __restrict__ W1,
                          const float* __restrict__ al, const float* __restrict__ ar,
                          float* __restrict__ z1, float* __restrict__ el,
                          float* __restrict__ er, int N) {
    int t = blockIdx.x * blockDim.x + threadIdx.x;      // (n, h) pairs, H=4
    if (t >= N * 4) return;
    int n = t >> 2, h = t & 3;
    float d = deg[n];
    float sl = 0.f, sr = 0.f;
    #pragma unroll 8
    for (int j = 0; j < 32; ++j) {
        float w = W1[h * 32 + j];
        float z = d * w;
        z1[n * 128 + h * 32 + j] = z;
        sl += z * al[h * 32 + j];
        sr += z * ar[h * 32 + j];
    }
    el[t] = sl; er[t] = sr;
}

// ---- edge attention: e = leaky_relu(el[src]+er[dst]); segment max ---------
template <int H>
__global__ void k_edge_max(const int* __restrict__ src, const int* __restrict__ dst,
                           const float* __restrict__ el, const float* __restrict__ er,
                           float* __restrict__ ebuf, unsigned* __restrict__ m, int E) {
    int e = blockIdx.x * blockDim.x + threadIdx.x;
    if (e >= E) return;
    int s = src[e], d = dst[e];
    #pragma unroll
    for (int h = 0; h < H; ++h) {
        float v = el[s * H + h] + er[d * H + h];
        v = v > 0.f ? v : 0.2f * v;                    // leaky_relu(0.2)
        ebuf[e * H + h] = v;
        atomicMax(&m[d * H + h], fmap(v));
    }
}

// ---- exp(e - max) and segment sum -----------------------------------------
template <int H>
__global__ void k_edge_exp(const int* __restrict__ dst, float* __restrict__ ebuf,
                           const unsigned* __restrict__ m, float* __restrict__ ssum, int E) {
    int e = blockIdx.x * blockDim.x + threadIdx.x;
    if (e >= E) return;
    int d = dst[e];
    #pragma unroll
    for (int h = 0; h < H; ++h) {
        float ex = expf(ebuf[e * H + h] - funmap(m[d * H + h]));
        ebuf[e * H + h] = ex;
        atomicAdd(&ssum[d * H + h], ex);
    }
}

__global__ void k_inv(const float* __restrict__ s, float* __restrict__ inv, int n) {
    int i = blockIdx.x * blockDim.x + threadIdx.x;
    if (i < n) inv[i] = s[i] > 0.f ? 1.0f / s[i] : 0.0f;
}

// ---- scatter: out[dst] += alpha * z[src], coalesced over feature dim ------
template <int H>
__global__ void k_scatter(const int* __restrict__ src, const int* __restrict__ dst,
                          const float* __restrict__ ebuf, const float* __restrict__ inv,
                          const float* __restrict__ z, float* __restrict__ out, int E) {
    constexpr int HD  = H * 32;
    constexpr int EPB = 256 / HD;                       // edges per 256-thread block
    int e = blockIdx.x * EPB + threadIdx.x / HD;
    int r = threadIdx.x % HD;                           // feature index within [H*32]
    if (e >= E) return;
    int s = src[e], d = dst[e];
    int h = r >> 5;
    float alpha = ebuf[e * H + h] * inv[d * H + h];
    atomicAdd(&out[d * HD + r], alpha * z[s * HD + r]);
}

// ---- h1 = relu(out1 + b1) -> f16 for WMMA ---------------------------------
__global__ void k_h1(const float* __restrict__ out1, const float* __restrict__ b1,
                     _Float16* __restrict__ h1h, int total) {
    int i = blockIdx.x * blockDim.x + threadIdx.x;
    if (i >= total) return;
    float v = fmaxf(out1[i] + b1[i & 127], 0.0f);
    h1h[i] = (_Float16)v;
}

__global__ void k_cvt(const float* __restrict__ a, _Float16* __restrict__ b, int n) {
    int i = blockIdx.x * blockDim.x + threadIdx.x;
    if (i < n) b[i] = (_Float16)a[i];
}

// convert [rows,10] f32 -> [rows,16] f16 zero-padded (keeps B rows 16B-aligned)
__global__ void k_cvt_pad10(const float* __restrict__ a, _Float16* __restrict__ b, int rows) {
    int i = blockIdx.x * blockDim.x + threadIdx.x;      // rows*16
    if (i >= rows * 16) return;
    int r = i >> 4, c = i & 15;
    b[i] = (c < 10) ? (_Float16)a[r * 10 + c] : (_Float16)0.0f;
}

// ---- el/er for layer 2 ----------------------------------------------------
__global__ void k_elr(const float* __restrict__ z, const float* __restrict__ al,
                      const float* __restrict__ ar, float* __restrict__ el,
                      float* __restrict__ er, int NH, int H) {
    int t = blockIdx.x * blockDim.x + threadIdx.x;
    if (t >= NH) return;
    int n = t / H, h = t % H;
    const float* zp = z + (long long)n * H * 32 + h * 32;
    float sl = 0.f, sr = 0.f;
    #pragma unroll 8
    for (int j = 0; j < 32; ++j) {
        sl += zp[j] * al[h * 32 + j];
        sr += zp[j] * ar[h * 32 + j];
    }
    el[t] = sl; er[t] = sr;
}

// ---- h2 = mean over 2 heads of relu(out2 + b2) ----------------------------
__global__ void k_h2(const float* __restrict__ out2, const float* __restrict__ b2,
                     float* __restrict__ h2, int N) {
    int t = blockIdx.x * blockDim.x + threadIdx.x;      // N*32
    if (t >= N * 32) return;
    int n = t >> 5, dd = t & 31;
    float a = fmaxf(out2[n * 64 + dd] + b2[dd], 0.0f);
    float b = fmaxf(out2[n * 64 + 32 + dd] + b2[32 + dd], 0.0f);
    h2[t] = 0.5f * (a + b);
}

// ---- per-graph mean readout -----------------------------------------------
__global__ void k_readout(const float* __restrict__ h2, const int* __restrict__ gid,
                          float* __restrict__ hg, float* __restrict__ counts, int N) {
    int t = blockIdx.x * blockDim.x + threadIdx.x;      // N*32
    if (t >= N * 32) return;
    int n = t >> 5, dd = t & 31;
    int g = gid[n];
    atomicAdd(&hg[g * 32 + dd], h2[t]);
    if (dd == 0) atomicAdd(&counts[g], 1.0f);
}

__global__ void k_readout_div(float* __restrict__ hg, const float* __restrict__ counts,
                              _Float16* __restrict__ hgh) {
    int t = blockIdx.x * blockDim.x + threadIdx.x;      // 64*32
    if (t >= NG * 32) return;
    float c = fmaxf(counts[t >> 5], 1.0f);
    float v = hg[t] / c;
    hg[t] = v;
    hgh[t] = (_Float16)v;
}

// ---- double-buffered LDS-staged WMMA GEMM ---------------------------------
// C[MxN] = act(A[MxK] @ B[KxN(ldb)] + bias).  A,B f16 row-major, K mult of 32,
// ldb mult of 8 (16B-aligned B rows).  Block = 256 threads = 8 waves -> 32x64
// macro-tile (2x4 16x16 WMMA tiles).  Staging is branch-free: A-row overreads
// (M not mult of 32) stay inside the workspace and only feed never-stored
// output rows.  One barrier per 32-K chunk; global loads for chunk c+1 are in
// flight while chunk c computes.  Fragment layouts per CDNA5 ISA 7.12.2.
#define GBM 32
#define GBN 64
#define BSTR 40                                        // ldsB col stride (halves)

__global__ __launch_bounds__(256)
void k_gemm_wmma(const _Float16* __restrict__ A, const _Float16* __restrict__ B,
                 const float* __restrict__ bias, float* __restrict__ Cf,
                 _Float16* __restrict__ Ch, int M, int K, int N, int ldb, int relu) {
    __shared__ _Float16 ldsA[2][GBM * 32];             // [row][k], stride 32 halves
    __shared__ _Float16 ldsB[2][GBN * BSTR];           // [col][k], stride 40 halves

    const int tid     = threadIdx.x;
    const int blocksN = (N + GBN - 1) / GBN;
    const int bm      = blockIdx.x / blocksN;
    const int bn      = blockIdx.x % blocksN;
    const int rowBase = bm * GBM;
    const int colBase = bn * GBN;

    const int wid  = tid >> 5, lane = tid & 31;
    const int lmod = lane & 15, lhalf = lane >> 4;
    const int wTm  = wid >> 2, wTn = wid & 3;          // 2x4 tile grid per block

    // A staging: thread -> row = tid>>3 (0..31), 4 halves at seg*4 (b64 load)
    const int aRow = tid >> 3, aSeg = tid & 7;
    const _Float16* ap = A + (long long)(rowBase + aRow) * K + aSeg * 4;
    const int ldsAoff = aRow * 32 + aSeg * 4;
    // B staging: thread -> k-row = tid>>3 (0..31), 8 cols at (tid&7)*8 (b128 load)
    const int bK = tid >> 3, bCseg = (tid & 7) * 8;
    const _Float16* bp = B + (long long)bK * ldb + colBase + bCseg;
    const int ldsBbase = bCseg * BSTR + bK;            // + i*BSTR immediate offsets

    const int arIdx = (wTm * 16 + lmod) * 32 + lhalf * 8;
    const int bcIdx = (wTn * 16 + lmod) * BSTR + lhalf * 8;
    const int nChunks = K >> 5;

    // prologue: stage chunk 0 into buffer 0
    v4h areg = *(const v4h*)ap;
    v8h breg = *(const v8h*)bp;
    *(v4h*)&ldsA[0][ldsAoff] = areg;
    #pragma unroll
    for (int i = 0; i < 8; ++i) ldsB[0][ldsBbase + i * BSTR] = breg[i];

    v8f acc = {};
    for (int c = 0; c < nChunks; ++c) {
        __syncthreads();                               // lds[c&1] ready for all
        const int buf = c & 1;
        if (c + 1 < nChunks) {                         // fetch next chunk early
            ap += 32;
            bp += (long long)32 * ldb;
            areg = *(const v4h*)ap;
            breg = *(const v8h*)bp;
        }
        v8h alo = *(const v8h*)&ldsA[buf][arIdx];
        v8h ahi = *(const v8h*)&ldsA[buf][arIdx + 16];
        v8h blo = *(const v8h*)&ldsB[buf][bcIdx];
        v8h bhi = *(const v8h*)&ldsB[buf][bcIdx + 16];
        v16h a, b;
        #pragma unroll
        for (int j = 0; j < 8; ++j) {
            a[j] = alo[j]; a[j + 8] = ahi[j];
            b[j] = blo[j]; b[j + 8] = bhi[j];
        }
        acc = __builtin_amdgcn_wmma_f32_16x16x32_f16(
            /*neg_a=*/false, a, /*neg_b=*/false, b,
            /*c_mod=*/(short)0, acc, /*reuse_a=*/false, /*reuse_b=*/false);
        if (c + 1 < nChunks) {                         // stage next into other buf
            *(v4h*)&ldsA[buf ^ 1][ldsAoff] = areg;
            #pragma unroll
            for (int i = 0; i < 8; ++i) ldsB[buf ^ 1][ldsBbase + i * BSTR] = breg[i];
        }
    }

    // ---- store (C layout: N=lane col, M split across lane halves) ----
    const int col = colBase + wTn * 16 + lmod;
    if (col < N) {
        float bv = bias ? bias[col] : 0.0f;
        #pragma unroll
        for (int r = 0; r < 8; ++r) {
            int row = rowBase + wTm * 16 + r + 8 * lhalf;
            if (row < M) {
                float v = acc[r] + bv;
                if (relu) v = fmaxf(v, 0.0f);
                if (Cf) Cf[(long long)row * N + col] = v;
                if (Ch) Ch[(long long)row * N + col] = (_Float16)v;
            }
        }
    }
}

// ---- row softmax over [64,10] ---------------------------------------------
__global__ void k_softmax(const float* __restrict__ logits, float* __restrict__ out) {
    int g = threadIdx.x;
    if (g >= NG) return;
    float mx = -3.0e38f;
    #pragma unroll
    for (int j = 0; j < 10; ++j) mx = fmaxf(mx, logits[g * 10 + j]);
    float e[10], s = 0.f;
    #pragma unroll
    for (int j = 0; j < 10; ++j) { e[j] = expf(logits[g * 10 + j] - mx); s += e[j]; }
    float inv = 1.0f / s;
    #pragma unroll
    for (int j = 0; j < 10; ++j) out[g * 10 + j] = e[j] * inv;
}

// ---------------------------------------------------------------------------
extern "C" void kernel_launch(void* const* d_in, const int* in_sizes, int n_in,
                              void* d_out, int out_size, void* d_ws, size_t ws_size,
                              hipStream_t stream) {
    (void)in_sizes; (void)n_in; (void)out_size; (void)ws_size;
    const int   N = NN, E = NE;
    const int*  src = (const int*)d_in[0];
    const int*  dst = (const int*)d_in[1];
    const int*  gid = (const int*)d_in[2];
    const float *W1 = (const float*)d_in[3],  *al1 = (const float*)d_in[4],
                *ar1 = (const float*)d_in[5], *b1  = (const float*)d_in[6],
                *W2 = (const float*)d_in[7],  *al2 = (const float*)d_in[8],
                *ar2 = (const float*)d_in[9], *b2  = (const float*)d_in[10],
                *cw1 = (const float*)d_in[11], *cb1 = (const float*)d_in[12],
                *cw2 = (const float*)d_in[13], *cb2 = (const float*)d_in[14],
                *cw3 = (const float*)d_in[15], *cb3 = (const float*)d_in[16],
                *cw4 = (const float*)d_in[17], *cb4 = (const float*)d_in[18],
                *cw5 = (const float*)d_in[19], *cb5 = (const float*)d_in[20];

    // bump allocator over d_ws (256B aligned)
    char*  base = (char*)d_ws;
    size_t off  = 0;
    auto alloc = [&](size_t bytes) -> void* {
        void* p = base + off;
        off = (off + bytes + 255) & ~(size_t)255;
        return p;
    };
    float*    deg  = (float*)alloc((size_t)N * 4);
    float*    z1   = (float*)alloc((size_t)N * 128 * 4);
    float*    el1  = (float*)alloc((size_t)N * 4 * 4);
    float*    er1  = (float*)alloc((size_t)N * 4 * 4);
    float*    e1   = (float*)alloc((size_t)E * 4 * 4);
    unsigned* m1   = (unsigned*)alloc((size_t)N * 4 * 4);
    float*    s1   = (float*)alloc((size_t)N * 4 * 4);
    float*    i1   = (float*)alloc((size_t)N * 4 * 4);
    float*    out1 = (float*)alloc((size_t)N * 128 * 4);
    _Float16* h1h  = (_Float16*)alloc((size_t)N * 128 * 2);
    _Float16* W2h  = (_Float16*)alloc(128 * 64 * 2);
    float*    z2   = (float*)alloc((size_t)N * 64 * 4);
    float*    el2  = (float*)alloc((size_t)N * 2 * 4);
    float*    er2  = (float*)alloc((size_t)N * 2 * 4);
    float*    e2   = (float*)alloc((size_t)E * 2 * 4);
    unsigned* m2   = (unsigned*)alloc((size_t)N * 2 * 4);
    float*    s2   = (float*)alloc((size_t)N * 2 * 4);
    float*    i2   = (float*)alloc((size_t)N * 2 * 4);
    float*    out2 = (float*)alloc((size_t)N * 64 * 4);
    float*    h2   = (float*)alloc((size_t)N * 32 * 4);
    float*    hg   = (float*)alloc(NG * 32 * 4);
    float*    cnt  = (float*)alloc(NG * 4);
    _Float16* hgh  = (_Float16*)alloc(NG * 32 * 2);
    _Float16* cw1h = (_Float16*)alloc(32 * 512 * 2);
    _Float16* cw2h = (_Float16*)alloc(512 * 1024 * 2);
    _Float16* cw3h = (_Float16*)alloc((size_t)1024 * 1024 * 2);
    _Float16* cw4h = (_Float16*)alloc(1024 * 512 * 2);
    _Float16* cw5h = (_Float16*)alloc(512 * 16 * 2);   // zero-padded to 16 cols
    _Float16* x1h  = (_Float16*)alloc(64 * 512 * 2);
    _Float16* x2h  = (_Float16*)alloc(64 * 1024 * 2);
    _Float16* x3h  = (_Float16*)alloc(64 * 1024 * 2);
    _Float16* x4h  = (_Float16*)alloc(64 * 512 * 2);
    float*    lg   = (float*)alloc(64 * 10 * 4);

    // zero accumulators (mapped-uint 0 acts as -inf for the max buffers)
    hipMemsetAsync(deg,  0, (size_t)N * 4, stream);
    hipMemsetAsync(m1,   0, (size_t)N * 4 * 4, stream);
    hipMemsetAsync(s1,   0, (size_t)N * 4 * 4, stream);
    hipMemsetAsync(out1, 0, (size_t)N * 128 * 4, stream);
    hipMemsetAsync(m2,   0, (size_t)N * 2 * 4, stream);
    hipMemsetAsync(s2,   0, (size_t)N * 2 * 4, stream);
    hipMemsetAsync(out2, 0, (size_t)N * 64 * 4, stream);
    hipMemsetAsync(hg,   0, NG * 32 * 4, stream);
    hipMemsetAsync(cnt,  0, NG * 4, stream);

    auto blks = [](long long n) { return (int)((n + 255) / 256); };
    auto gemm = [&](const _Float16* A, const _Float16* B, const float* bias,
                    float* Cf, _Float16* Ch, int M, int K, int Nc, int ldb, int relu) {
        int nblocks = ((M + GBM - 1) / GBM) * ((Nc + GBN - 1) / GBN);
        k_gemm_wmma<<<nblocks, 256, 0, stream>>>(A, B, bias, Cf, Ch, M, K, Nc, ldb, relu);
    };

    // node feature + GAT layer 1
    k_degree<<<blks(E), 256, 0, stream>>>(dst, deg, E);
    k_l1_node<<<blks((long long)N * 4), 256, 0, stream>>>(deg, W1, al1, ar1, z1, el1, er1, N);
    k_edge_max<4><<<blks(E), 256, 0, stream>>>(src, dst, el1, er1, e1, m1, E);
    k_edge_exp<4><<<blks(E), 256, 0, stream>>>(dst, e1, m1, s1, E);
    k_inv<<<blks((long long)N * 4), 256, 0, stream>>>(s1, i1, N * 4);
    k_scatter<4><<<(E + 1) / 2, 256, 0, stream>>>(src, dst, e1, i1, z1, out1, E);
    k_h1<<<blks((long long)N * 128), 256, 0, stream>>>(out1, b1, h1h, N * 128);

    // GAT layer 2: z2 = h1 @ W2 via WMMA
    k_cvt<<<blks(128 * 64), 256, 0, stream>>>(W2, W2h, 128 * 64);
    gemm(h1h, W2h, nullptr, z2, nullptr, N, 128, 64, 64, 0);
    k_elr<<<blks((long long)N * 2), 256, 0, stream>>>(z2, al2, ar2, el2, er2, N * 2, 2);
    k_edge_max<2><<<blks(E), 256, 0, stream>>>(src, dst, el2, er2, e2, m2, E);
    k_edge_exp<2><<<blks(E), 256, 0, stream>>>(dst, e2, m2, s2, E);
    k_inv<<<blks((long long)N * 2), 256, 0, stream>>>(s2, i2, N * 2);
    k_scatter<2><<<(E + 3) / 4, 256, 0, stream>>>(src, dst, e2, i2, z2, out2, E);
    k_h2<<<blks((long long)N * 32), 256, 0, stream>>>(out2, b2, h2, N);

    // graph mean readout
    k_readout<<<blks((long long)N * 32), 256, 0, stream>>>(h2, gid, hg, cnt, N);
    k_readout_div<<<8, 256, 0, stream>>>(hg, cnt, hgh);

    // MLP classifier (all WMMA)
    k_cvt<<<blks(32 * 512), 256, 0, stream>>>(cw1, cw1h, 32 * 512);
    k_cvt<<<blks(512 * 1024), 256, 0, stream>>>(cw2, cw2h, 512 * 1024);
    k_cvt<<<blks(1024 * 1024), 256, 0, stream>>>(cw3, cw3h, 1024 * 1024);
    k_cvt<<<blks(1024 * 512), 256, 0, stream>>>(cw4, cw4h, 1024 * 512);
    k_cvt_pad10<<<blks(512 * 16), 256, 0, stream>>>(cw5, cw5h, 512);
    gemm(hgh, cw1h, cb1, nullptr, x1h, 64, 32, 512, 512, 1);
    gemm(x1h, cw2h, cb2, nullptr, x2h, 64, 512, 1024, 1024, 1);
    gemm(x2h, cw3h, cb3, nullptr, x3h, 64, 1024, 1024, 1024, 1);
    gemm(x3h, cw4h, cb4, nullptr, x4h, 64, 1024, 512, 512, 1);
    gemm(x4h, cw5h, cb5, lg, nullptr, 64, 512, 10, 16, 0);
    k_softmax<<<1, 64, 0, stream>>>(lg, (float*)d_out);
}
